// MultiTaskVideoClassifier_29480655520209
// MI455X (gfx1250) — matrix operand order repaired and checked by reading, not compiled, not run on hardware
//
#include <hip/hip_runtime.h>

// ---------------------------------------------------------------------------
// MI455X (gfx1250): all large GEMMs via v_wmma_f32_16x16x32_bf16 (wave32,
// fp32 accumulate), double-buffered LDS staging with GLOBAL_LOAD_ASYNC_TO_LDS
// (ASYNCcnt) for the A tiles.  Fused epilogues for bias / pos-enc / residual /
// relu / MoE combine-scale accumulation.  Attention is a dedicated WMMA kernel.
// ---------------------------------------------------------------------------

typedef __attribute__((ext_vector_type(16))) __bf16 bf16x16;
typedef __attribute__((ext_vector_type(8)))  float  f32x8;

#define D_MODEL 1024
#define NHEAD   8
#define NLAYER  4
#define NEXP    4
#define FFDIM   2048
#define VISD    1536
#define SEQ     128
#define BATCH   64
#define NTOK    (BATCH * SEQ)   // 8192 tokens

// ---------------------------- small helpers --------------------------------

__device__ __forceinline__ float wave_reduce_sum(float v) {
#pragma unroll
  for (int o = 16; o > 0; o >>= 1) v += __shfl_xor(v, o, 32);
  return v;
}

__device__ __forceinline__ void async_ldst_b128(const void* gaddr, unsigned ldsoff) {
  // GLOBAL_LOAD_ASYNC_TO_LDS_B128 (GV mode): per-lane 16B global -> LDS copy,
  // tracked by ASYNCcnt (CDNA5 ISA).
  asm volatile("global_load_async_to_lds_b128 %0, %1, off"
               :
               : "v"(ldsoff), "v"((unsigned long long)(size_t)gaddr)
               : "memory");
}

__device__ __forceinline__ void wait_async_all() {
  asm volatile("s_wait_asynccnt 0x0" ::: "memory");
}

__global__ __launch_bounds__(256)
void f32_to_bf16_kern(const float* __restrict__ in, __bf16* __restrict__ out, int n) {
  int i = blockIdx.x * 256 + threadIdx.x;
  if (i < n) out[i] = (__bf16)in[i];
}

__global__ __launch_bounds__(256)
void pe_kern(float* __restrict__ pe) {   // SEQ x D_MODEL sinusoidal table
  int i = blockIdx.x * 256 + threadIdx.x;
  int s = i >> 10, d = i & 1023;
  int k = d >> 1;
  float div = __expf(-(float)(2 * k) * (9.210340371976184f / 1024.f)); // ln(1e4)/D
  pe[i] = (d & 1) ? __cosf((float)s * div) : __sinf((float)s * div);
}

// ------------------------------ GEMM kernel --------------------------------
// C[M,N] = A[M,K](bf16) @ B[K,N](bf16,row-major) with fused epilogues.
// Block: 256 thr = 8 waves.  Tile 128x128, K-step 32, double-buffered LDS.
// Wave grid 2x4, each wave -> 64x32 region = 4x2 wmma 16x16 subtiles.
// A tiles streamed global->LDS asynchronously; B tiles transposed via VGPRs.

#define GBK   32
#define GPAD  8
#define GLDS  (GBK + GPAD)   // 40 elements -> 80B row stride (16B aligned)

enum { MODE_BIAS = 0, MODE_PE = 1, MODE_RES = 2, MODE_RELU = 3, MODE_SCALE = 4 };

template <typename OutT, int MODE>
__global__ __launch_bounds__(256)
void gemm_bf16_kern(const __bf16* __restrict__ A, const __bf16* __restrict__ B,
                    const float* __restrict__ bias, const float* __restrict__ extra,
                    OutT* __restrict__ C, int M, int N, int K) {
  __shared__ __bf16 As[2][128 * GLDS];
  __shared__ __bf16 Bs[2][128 * GLDS];   // stored transposed: Bs[n][k]

  const int tid  = threadIdx.x;
  const int lane = tid & 31;
  const int wave = tid >> 5;
  const int wm = wave >> 2, wn = wave & 3;
  const int nBase = blockIdx.x * 128, mBase = blockIdx.y * 128;

  f32x8 acc[4][2];
#pragma unroll
  for (int i = 0; i < 4; i++)
#pragma unroll
    for (int j = 0; j < 2; j++) acc[i][j] = (f32x8)0.0f;

  const int khA = (lane < 16) ? 0 : 16;   // 16-bit A frag K-halves (contiguous k order, same as B)
  const int khB = (lane < 16) ? 0 : 16;   // 16-bit B frag K-half
  const int mrow = lane & 15;

  // --- staging helpers (2 chunks of 16B per thread per tile) ---
  auto stageA = [&](int buf, int k0) {
#pragma unroll
    for (int c = tid; c < 512; c += 256) {
      int r = c >> 2, kc = (c & 3) << 3;
      const __bf16* src = A + (size_t)(mBase + r) * K + k0 + kc;
      unsigned lo = (unsigned)(size_t)&As[buf][r * GLDS + kc];
      async_ldst_b128(src, lo);
    }
  };
  auto stageB = [&](int buf, int k0) {
#pragma unroll
    for (int c = tid; c < 512; c += 256) {
      int kr = c >> 4, nc = (c & 15) << 3;
      const __bf16* src = B + (size_t)(k0 + kr) * N + nBase + nc;
#pragma unroll
      for (int j = 0; j < 8; j++) Bs[buf][(nc + j) * GLDS + kr] = src[j];
    }
  };

  stageA(0, 0);
  stageB(0, 0);
  wait_async_all();
  __syncthreads();

  const int ksteps = K / GBK;
  for (int ks = 0; ks < ksteps; ks++) {
    const int cur = ks & 1, nxt = cur ^ 1;
    if (ks + 1 < ksteps) {        // prefetch next tile while computing
      stageA(nxt, (ks + 1) * GBK);
      stageB(nxt, (ks + 1) * GBK);
    }

    bf16x16 afrag[4], bfrag[2];
#pragma unroll
    for (int mi = 0; mi < 4; mi++) {
      const __bf16* p = &As[cur][(wm * 64 + mi * 16 + mrow) * GLDS + khA];
#pragma unroll
      for (int j = 0; j < 8; j++) { afrag[mi][j] = p[j]; afrag[mi][8 + j] = p[8 + j]; }
    }
#pragma unroll
    for (int ni = 0; ni < 2; ni++) {
      const __bf16* p = &Bs[cur][(wn * 32 + ni * 16 + mrow) * GLDS + khB];
#pragma unroll
      for (int j = 0; j < 16; j++) bfrag[ni][j] = p[j];
    }
#pragma unroll
    for (int mi = 0; mi < 4; mi++)
#pragma unroll
      for (int ni = 0; ni < 2; ni++)
        acc[mi][ni] = __builtin_amdgcn_wmma_f32_16x16x32_bf16(
            false, afrag[mi], false, bfrag[ni], (short)0, acc[mi][ni], false, false);

    wait_async_all();   // next tile's async copies landed
    __syncthreads();    // everyone done reading `cur`, `nxt` fully staged
  }

  // Epilogue.  D layout: VGPR v -> row v + (lane<16?0:8); col = lane&15.
#pragma unroll
  for (int mi = 0; mi < 4; mi++) {
    int rbase = mBase + wm * 64 + mi * 16 + ((lane < 16) ? 0 : 8);
#pragma unroll
    for (int ni = 0; ni < 2; ni++) {
      int col = nBase + wn * 32 + ni * 16 + (lane & 15);
#pragma unroll
      for (int v = 0; v < 8; v++) {
        int r = rbase + v;
        float val = acc[mi][ni][v] + bias[col];
        if (MODE == MODE_PE)   val += extra[(r & (SEQ - 1)) * D_MODEL + col];
        if (MODE == MODE_RES)  val += extra[(size_t)r * N + col];
        if (MODE == MODE_RELU) val = fmaxf(val, 0.f);
        if (MODE == MODE_SCALE) {
          ((float*)C)[(size_t)r * N + col] += extra[r * NEXP] * val;
        } else {
          C[(size_t)r * N + col] = (OutT)val;
        }
      }
    }
  }
}

// --------------------------- attention kernel ------------------------------
// One block per (b, h, query-tile-of-32).  scores = q k^T / sqrt(HD), softmax,
// o = probs @ v.  q,k fragments gathered from the row-major bf16 QKV buffer
// (layout matches A / B WMMA fragments directly); v staged transposed in LDS.

#define APAD 8
#define ALD  (128 + APAD)   // 136 -> 272B stride, 16B aligned

__global__ __launch_bounds__(256)
void attention_kern(const __bf16* __restrict__ qkv, __bf16* __restrict__ out) {
  const int qt = blockIdx.x & 3;
  const int bh = blockIdx.x >> 2;
  const int b = bh >> 3, h = bh & 7;
  const int tid = threadIdx.x, lane = tid & 31, wave = tid >> 5;
  const int wm = wave >> 2, wn = wave & 3;   // wm: 2x16 rows, wn: 4x32 cols

  __shared__ __bf16 vT[128 * ALD];   // v transposed: vT[d][s]
  __shared__ float  sc[32 * ALD];    // fp32 scores for this q-tile
  __shared__ __bf16 pb[32 * ALD];    // bf16 softmax probs

  const size_t rs = 3 * D_MODEL;     // qkv row stride
  const __bf16* qb = qkv + (size_t)b * SEQ * rs + h * 128;
  const __bf16* kb = qb + D_MODEL;
  const __bf16* vb = qb + 2 * D_MODEL;

  // stage v transposed (128x128)
#pragma unroll
  for (int c = tid; c < 2048; c += 256) {
    int s = c >> 4, dc = (c & 15) << 3;
    const __bf16* src = vb + (size_t)s * rs + dc;
#pragma unroll
    for (int j = 0; j < 8; j++) vT[(dc + j) * ALD + s] = src[j];
  }
  __syncthreads();

  const int khA = (lane < 16) ? 0 : 16;
  const int khB = (lane < 16) ? 0 : 16;
  const int mrow = lane & 15;

  // ---- scores = q @ k^T  (32 x 128, K = 128) ----
  f32x8 acc[2];
  acc[0] = (f32x8)0.0f; acc[1] = (f32x8)0.0f;
  for (int k0 = 0; k0 < 128; k0 += 32) {
    bf16x16 a, bf[2];
    const __bf16* pa = qb + (size_t)(qt * 32 + wm * 16 + mrow) * rs + k0 + khA;
#pragma unroll
    for (int j = 0; j < 8; j++) { a[j] = pa[j]; a[8 + j] = pa[8 + j]; }
#pragma unroll
    for (int ni = 0; ni < 2; ni++) {
      const __bf16* pk = kb + (size_t)(wn * 32 + ni * 16 + mrow) * rs + k0 + khB;
#pragma unroll
      for (int j = 0; j < 16; j++) bf[ni][j] = pk[j];
      acc[ni] = __builtin_amdgcn_wmma_f32_16x16x32_bf16(
          false, a, false, bf[ni], (short)0, acc[ni], false, false);
    }
  }
  const float scale = 0.08838834764831845f;   // 1/sqrt(128)
#pragma unroll
  for (int ni = 0; ni < 2; ni++) {
    int rbase = wm * 16 + ((lane < 16) ? 0 : 8);
    int col = wn * 32 + ni * 16 + mrow;
#pragma unroll
    for (int v = 0; v < 8; v++) sc[(rbase + v) * ALD + col] = acc[ni][v] * scale;
  }
  __syncthreads();

  // ---- row softmax: 8 lanes per row, 16 cols each (stays inside a wave) ----
  {
    int r = tid >> 3;
    int cg = (tid & 7) * 16;
    float mx = -1e30f;
#pragma unroll
    for (int j = 0; j < 16; j++) mx = fmaxf(mx, sc[r * ALD + cg + j]);
#pragma unroll
    for (int o = 1; o < 8; o <<= 1) mx = fmaxf(mx, __shfl_xor(mx, o, 32));
    float e[16], su = 0.f;
#pragma unroll
    for (int j = 0; j < 16; j++) { e[j] = __expf(sc[r * ALD + cg + j] - mx); su += e[j]; }
#pragma unroll
    for (int o = 1; o < 8; o <<= 1) su += __shfl_xor(su, o, 32);
    float inv = 1.f / su;
#pragma unroll
    for (int j = 0; j < 16; j++) pb[r * ALD + cg + j] = (__bf16)(e[j] * inv);
  }
  __syncthreads();

  // ---- o = probs @ v  (32 x 128, K = 128) ----
  f32x8 acc2[2];
  acc2[0] = (f32x8)0.0f; acc2[1] = (f32x8)0.0f;
  for (int k0 = 0; k0 < 128; k0 += 32) {
    bf16x16 a, bf[2];
    const __bf16* pa = &pb[(wm * 16 + mrow) * ALD + k0 + khA];
#pragma unroll
    for (int j = 0; j < 8; j++) { a[j] = pa[j]; a[8 + j] = pa[8 + j]; }
#pragma unroll
    for (int ni = 0; ni < 2; ni++) {
      const __bf16* pv = &vT[(wn * 32 + ni * 16 + mrow) * ALD + k0 + khB];
#pragma unroll
      for (int j = 0; j < 16; j++) bf[ni][j] = pv[j];
      acc2[ni] = __builtin_amdgcn_wmma_f32_16x16x32_bf16(
          false, a, false, bf[ni], (short)0, acc2[ni], false, false);
    }
  }
#pragma unroll
  for (int ni = 0; ni < 2; ni++) {
    int rbase = b * SEQ + qt * 32 + wm * 16 + ((lane < 16) ? 0 : 8);
    int col = h * 128 + wn * 32 + ni * 16 + mrow;
#pragma unroll
    for (int v = 0; v < 8; v++)
      out[(size_t)(rbase + v) * D_MODEL + col] = (__bf16)acc2[ni][v];
  }
}

// -------------------- residual + layernorm (fp32 + bf16 out) ---------------

__global__ __launch_bounds__(256)
void layernorm_kern(const float* __restrict__ in, const float* __restrict__ res,
                    const float* __restrict__ g, const float* __restrict__ be,
                    float* __restrict__ xo, __bf16* __restrict__ xb) {
  int row = blockIdx.x, tid = threadIdx.x;
  __shared__ float buf[1024];
  __shared__ float redS[8], redS2[8];
  float s = 0.f, s2 = 0.f;
#pragma unroll
  for (int i = 0; i < 4; i++) {
    int d = tid + i * 256;
    float y = in[(size_t)row * 1024 + d];
    if (res) y += res[(size_t)row * 1024 + d];
    buf[d] = y; s += y; s2 += y * y;
  }
  s = wave_reduce_sum(s); s2 = wave_reduce_sum(s2);
  if ((tid & 31) == 0) { redS[tid >> 5] = s; redS2[tid >> 5] = s2; }
  __syncthreads();
  float ts = 0.f, ts2 = 0.f;
#pragma unroll
  for (int w = 0; w < 8; w++) { ts += redS[w]; ts2 += redS2[w]; }
  float m = ts * (1.f / 1024.f);
  float v = ts2 * (1.f / 1024.f) - m * m;
  float rstd = rsqrtf(v + 1e-5f);
#pragma unroll
  for (int i = 0; i < 4; i++) {
    int d = tid + i * 256;
    float yn = (buf[d] - m) * rstd * g[d] + be[d];
    xo[(size_t)row * 1024 + d] = yn;
    xb[(size_t)row * 1024 + d] = (__bf16)yn;
  }
}

// ------------------ router: top-2 softmax combine weights ------------------

__global__ __launch_bounds__(256)
void router_kern(const float* __restrict__ x, const float* __restrict__ rw,
                 const float* __restrict__ rb, float* __restrict__ combine) {
  int n = blockIdx.x * 8 + (threadIdx.x >> 5);
  int lane = threadIdx.x & 31;
  float acc[NEXP] = {0.f, 0.f, 0.f, 0.f};
  for (int d = lane; d < D_MODEL; d += 32) {
    float xv = x[(size_t)n * D_MODEL + d];
#pragma unroll
    for (int e = 0; e < NEXP; e++) acc[e] += xv * rw[d * NEXP + e];
  }
#pragma unroll
  for (int e = 0; e < NEXP; e++) acc[e] = wave_reduce_sum(acc[e]);
  if (lane == 0) {
    float l[NEXP];
#pragma unroll
    for (int e = 0; e < NEXP; e++) l[e] = acc[e] + rb[e];
    int i1 = 0;
    for (int e = 1; e < NEXP; e++) if (l[e] > l[i1]) i1 = e;
    int i2 = -1;
    for (int e = 0; e < NEXP; e++)
      if (e != i1 && (i2 < 0 || l[e] > l[i2])) i2 = e;
    float e2 = __expf(l[i2] - l[i1]);
    float p1 = 1.f / (1.f + e2), p2 = e2 / (1.f + e2);
    float c[NEXP] = {0.f, 0.f, 0.f, 0.f};
    c[i1] = p1; c[i2] = p2;
#pragma unroll
    for (int e = 0; e < NEXP; e++) combine[n * NEXP + e] = c[e];
  }
}

// ------------- attention pooling over time + 4 task heads ------------------

__global__ __launch_bounds__(256)
void pool_heads_kern(const float* __restrict__ x, const float* __restrict__ aw,
                     const float* __restrict__ ab, const float* __restrict__ hw,
                     const float* __restrict__ hb, float* __restrict__ out) {
  int b = blockIdx.x, tid = threadIdx.x, lane = tid & 31, wave = tid >> 5;
  __shared__ float sco[SEQ];
  __shared__ float pooled[D_MODEL];
  __shared__ float ssum;
  for (int i = 0; i < 16; i++) {
    int s = wave * 16 + i;
    float a = 0.f;
    for (int d = lane; d < D_MODEL; d += 32)
      a += x[((size_t)b * SEQ + s) * D_MODEL + d] * aw[d];
    a = wave_reduce_sum(a);
    if (lane == 0) sco[s] = a + ab[0];
  }
  __syncthreads();
  if (tid == 0) {
    float m = -1e30f;
    for (int s = 0; s < SEQ; s++) m = fmaxf(m, sco[s]);
    float su = 0.f;
    for (int s = 0; s < SEQ; s++) { float e = __expf(sco[s] - m); sco[s] = e; su += e; }
    ssum = su;
  }
  __syncthreads();
  float inv = 1.f / ssum;
  for (int d = tid; d < D_MODEL; d += 256) {
    float a = 0.f;
    for (int s = 0; s < SEQ; s++)
      a += sco[s] * x[((size_t)b * SEQ + s) * D_MODEL + d];
    pooled[d] = a * inv;
  }
  __syncthreads();
  // 16 head outputs (t,c): 2 per wave
#pragma unroll
  for (int j = 0; j < 2; j++) {
    int oo = wave * 2 + j, t = oo >> 2, c = oo & 3;
    float a = 0.f;
    for (int d = lane; d < D_MODEL; d += 32)
      a += pooled[d] * hw[((size_t)t * D_MODEL + d) * 4 + c];
    a = wave_reduce_sum(a);
    if (lane == 0) out[t * (BATCH * 4) + b * 4 + c] = a + hb[t * 4 + c];
  }
}

// ------------------------------- launcher ----------------------------------

extern "C" void kernel_launch(void* const* d_in, const int* in_sizes, int n_in,
                              void* d_out, int out_size, void* d_ws, size_t ws_size,
                              hipStream_t stream) {
  const float* vis   = (const float*)d_in[0];
  const float* projw = (const float*)d_in[1];
  const float* projb = (const float*)d_in[2];
  const float* inw   = (const float*)d_in[3];
  const float* inb   = (const float*)d_in[4];
  const float* outw  = (const float*)d_in[5];
  const float* outb  = (const float*)d_in[6];
  const float* ln1g  = (const float*)d_in[7];
  const float* ln1b  = (const float*)d_in[8];
  const float* ln2g  = (const float*)d_in[9];
  const float* ln2b  = (const float*)d_in[10];
  const float* rw    = (const float*)d_in[11];
  const float* rb    = (const float*)d_in[12];
  const float* ew1   = (const float*)d_in[13];
  const float* eb1   = (const float*)d_in[14];
  const float* ew2   = (const float*)d_in[15];
  const float* eb2   = (const float*)d_in[16];
  const float* attw  = (const float*)d_in[17];
  const float* attb  = (const float*)d_in[18];
  const float* hw    = (const float*)d_in[19];
  const float* hb    = (const float*)d_in[20];
  (void)in_sizes; (void)n_in; (void)out_size; (void)ws_size;

  // bump-allocate workspace
  char* p = (char*)d_ws;
  auto alloc = [&](size_t bytes) {
    void* r = (void*)p;
    p += (bytes + 255) & ~(size_t)255;
    return r;
  };
  __bf16* visb  = (__bf16*)alloc((size_t)NTOK * VISD * 2);
  __bf16* projw_b = (__bf16*)alloc((size_t)VISD * D_MODEL * 2);
  __bf16* inw_b = (__bf16*)alloc((size_t)NLAYER * D_MODEL * 3 * D_MODEL * 2);
  __bf16* outw_b = (__bf16*)alloc((size_t)NLAYER * D_MODEL * D_MODEL * 2);
  __bf16* e1w_b = (__bf16*)alloc((size_t)NLAYER * NEXP * D_MODEL * FFDIM * 2);
  __bf16* e2w_b = (__bf16*)alloc((size_t)NLAYER * NEXP * FFDIM * D_MODEL * 2);
  float*  pe    = (float*)alloc((size_t)SEQ * D_MODEL * 4);
  float*  x     = (float*)alloc((size_t)NTOK * D_MODEL * 4);
  __bf16* xb    = (__bf16*)alloc((size_t)NTOK * D_MODEL * 2);
  float*  tmp   = (float*)alloc((size_t)NTOK * D_MODEL * 4);
  __bf16* qkvb  = (__bf16*)alloc((size_t)NTOK * 3 * D_MODEL * 2);
  __bf16* attnb = (__bf16*)alloc((size_t)NTOK * D_MODEL * 2);
  __bf16* hbuf  = (__bf16*)alloc((size_t)NTOK * FFDIM * 2);
  float*  moe   = (float*)alloc((size_t)NTOK * D_MODEL * 4);
  float*  comb  = (float*)alloc((size_t)NTOK * NEXP * 4);

  auto conv = [&](const float* src, __bf16* dst, int n) {
    f32_to_bf16_kern<<<(n + 255) / 256, 256, 0, stream>>>(src, dst, n);
  };

  // weight / input conversions (recomputed each call: deterministic, cheap)
  conv(vis,   visb,   NTOK * VISD);
  conv(projw, projw_b, VISD * D_MODEL);
  conv(inw,   inw_b,  NLAYER * D_MODEL * 3 * D_MODEL);
  conv(outw,  outw_b, NLAYER * D_MODEL * D_MODEL);
  conv(ew1,   e1w_b,  NLAYER * NEXP * D_MODEL * FFDIM);
  conv(ew2,   e2w_b,  NLAYER * NEXP * FFDIM * D_MODEL);
  pe_kern<<<(SEQ * D_MODEL) / 256, 256, 0, stream>>>(pe);

  // x = vis @ proj_w + proj_b + pos_enc
  gemm_bf16_kern<float, MODE_PE><<<dim3(D_MODEL / 128, NTOK / 128), 256, 0, stream>>>(
      visb, projw_b, projb, pe, x, NTOK, D_MODEL, VISD);
  conv(x, xb, NTOK * D_MODEL);

  for (int l = 0; l < NLAYER; l++) {
    // qkv = x @ wi + bi   -> bf16
    gemm_bf16_kern<__bf16, MODE_BIAS><<<dim3(3 * D_MODEL / 128, NTOK / 128), 256, 0, stream>>>(
        xb, inw_b + (size_t)l * D_MODEL * 3 * D_MODEL, inb + l * 3 * D_MODEL,
        nullptr, qkvb, NTOK, 3 * D_MODEL, D_MODEL);
    // attention -> bf16
    attention_kern<<<BATCH * NHEAD * 4, 256, 0, stream>>>(qkvb, attnb);
    // out-proj + residual
    gemm_bf16_kern<float, MODE_RES><<<dim3(D_MODEL / 128, NTOK / 128), 256, 0, stream>>>(
        attnb, outw_b + (size_t)l * D_MODEL * D_MODEL, outb + l * D_MODEL,
        x, tmp, NTOK, D_MODEL, D_MODEL);
    // LN1 -> x (fp32) + xb (bf16)
    layernorm_kern<<<NTOK, 256, 0, stream>>>(tmp, nullptr, ln1g + l * D_MODEL,
                                             ln1b + l * D_MODEL, x, xb);
    // router combine weights
    router_kern<<<NTOK / 8, 256, 0, stream>>>(x, rw + (size_t)l * D_MODEL * NEXP,
                                              rb + l * NEXP, comb);
    // MoE (dense over experts, matching reference)
    hipMemsetAsync(moe, 0, (size_t)NTOK * D_MODEL * 4, stream);
    for (int e = 0; e < NEXP; e++) {
      const __bf16* w1 = e1w_b + ((size_t)l * NEXP + e) * D_MODEL * FFDIM;
      const __bf16* w2 = e2w_b + ((size_t)l * NEXP + e) * FFDIM * D_MODEL;
      gemm_bf16_kern<__bf16, MODE_RELU><<<dim3(FFDIM / 128, NTOK / 128), 256, 0, stream>>>(
          xb, w1, eb1 + ((size_t)l * NEXP + e) * FFDIM, nullptr, hbuf,
          NTOK, FFDIM, D_MODEL);
      gemm_bf16_kern<float, MODE_SCALE><<<dim3(D_MODEL / 128, NTOK / 128), 256, 0, stream>>>(
          hbuf, w2, eb2 + ((size_t)l * NEXP + e) * D_MODEL, comb + e, moe,
          NTOK, D_MODEL, FFDIM);
    }
    // LN2 over (x + moe) -> x, xb
    layernorm_kern<<<NTOK, 256, 0, stream>>>(moe, x, ln2g + l * D_MODEL,
                                             ln2b + l * D_MODEL, x, xb);
  }

  // attention pooling + task heads -> d_out [4, 64, 4] fp32
  pool_heads_kern<<<BATCH, 256, 0, stream>>>(x, attw, attb, hw, hb, (float*)d_out);
}
